// HGT_64587718197897
// MI455X (gfx1250) — matrix operand order repaired and checked by reading, not compiled, not run on hardware
//
#include <hip/hip_runtime.h>
#include <hip/hip_bf16.h>
#include <float.h>

#define HID 128
#define NH  8
#define HD  16

typedef __attribute__((ext_vector_type(16))) __bf16 v16bf;
typedef __attribute__((ext_vector_type(8)))  float  v8f;

// ---------- helpers ----------

__device__ __forceinline__ unsigned int f2bf1(float f) {
  // round-to-nearest-even f32 -> bf16 bits
  unsigned int u = __float_as_uint(f);
  u += 0x7FFFu + ((u >> 16) & 1u);
  return u >> 16;
}
__device__ __forceinline__ unsigned int pack_bf2(float lo, float hi) {
  return f2bf1(lo) | (f2bf1(hi) << 16);
}

union Frag16 {
  unsigned int u[8];
  v16bf v;
};

__device__ __forceinline__ void atomicMaxFloat(float* addr, float v) {
  // standard IEEE int-punning max (addr initialized to -FLT_MAX)
  if (v >= 0.f) atomicMax((int*)addr, __float_as_int(v));
  else          atomicMin((unsigned int*)addr, __float_as_uint(v));
}

__device__ __forceinline__ void load16f(const float* p, float* o) {
  const float4* q = (const float4*)p;
  float4 a = q[0], b = q[1], c = q[2], d = q[3];
  o[0]=a.x; o[1]=a.y; o[2]=a.z; o[3]=a.w;
  o[4]=b.x; o[5]=b.y; o[6]=b.z; o[7]=b.w;
  o[8]=c.x; o[9]=c.y; o[10]=c.z; o[11]=c.w;
  o[12]=d.x; o[13]=d.y; o[14]=d.z; o[15]=d.w;
}

// ---------- WMMA GEMM:  C[M,128] = epi( A[M,K] @ W[K,128] + bias ) ----------
// EPI 0: none   EPI 1: relu   EPI 2: beta*val + (1-beta)*xold  (beta=sigmoid(skip))
// Block = 256 threads = 8 waves; block covers 128 rows; wave w covers rows [w*16,w*16+16).
// W is packed once per block into LDS in per-lane bf16 B-fragment layout.

template<int K, int EPI>
__global__ __launch_bounds__(256)
void gemm_wmma_kernel(const float* __restrict__ A,
                      const float* __restrict__ W,
                      const float* __restrict__ bias,
                      const float* __restrict__ xold,
                      const float* __restrict__ skipv,
                      float* __restrict__ C,
                      int M)
{
  constexpr int KSTEPS = K / 32;
  __shared__ unsigned int sB[KSTEPS * 8 * 32 * 8];  // [kstep][ntile][lane][8 dwords]

  const int tid = threadIdx.x;

  // --- pack W (f32 row-major [K,128]) into WMMA B fragments (bf16) in LDS ---
  for (int slot = tid; slot < KSTEPS * 8 * 32; slot += 256) {
    const int plane = slot & 31;
    const int nt    = (slot >> 5) & 7;
    const int ks    = slot >> 8;
    const int n     = nt * 16 + (plane & 15);
    const int phalf = plane >> 4;
    unsigned int* dst = &sB[slot * 8];
#pragma unroll
    for (int j = 0; j < 8; ++j) {
      const int k0 = ks * 32 + (j < 4 ? phalf * 8 + j * 2 : 16 + phalf * 8 + (j - 4) * 2);
      dst[j] = pack_bf2(W[(size_t)k0 * HID + n], W[(size_t)(k0 + 1) * HID + n]);
    }
  }
  __syncthreads();

  const int wave    = tid >> 5;
  const int lane    = tid & 31;
  const int half    = lane >> 4;
  const int rowBase = blockIdx.x * 128 + wave * 16;
  const int m       = rowBase + (lane & 15);
  const bool valid  = (m < M);
  const float* ap   = A + (size_t)(valid ? m : 0) * K;  // clamp OOB lanes to row 0

  v8f acc[8] = {};

  for (int ks = 0; ks < KSTEPS; ++ks) {
    // A fragment: lane holds row m, K = ks*32 + {half*8..+7, 16+half*8..+7}
    Frag16 fa;
    {
      const float* p0 = ap + ks * 32 + half * 8;
      const float* p1 = ap + ks * 32 + 16 + half * 8;
      float4 f0 = ((const float4*)p0)[0];
      float4 f1 = ((const float4*)p0)[1];
      float4 f2 = ((const float4*)p1)[0];
      float4 f3 = ((const float4*)p1)[1];
      fa.u[0] = pack_bf2(f0.x, f0.y);
      fa.u[1] = pack_bf2(f0.z, f0.w);
      fa.u[2] = pack_bf2(f1.x, f1.y);
      fa.u[3] = pack_bf2(f1.z, f1.w);
      fa.u[4] = pack_bf2(f2.x, f2.y);
      fa.u[5] = pack_bf2(f2.z, f2.w);
      fa.u[6] = pack_bf2(f3.x, f3.y);
      fa.u[7] = pack_bf2(f3.z, f3.w);
    }
    if (ks + 1 < KSTEPS) __builtin_prefetch(ap + (ks + 1) * 32, 0, 0);

#pragma unroll
    for (int nt = 0; nt < 8; ++nt) {
      Frag16 fb;
      const unsigned int* src = &sB[((ks * 8 + nt) * 32 + lane) * 8];
#pragma unroll
      for (int j = 0; j < 8; ++j) fb.u[j] = src[j];
      acc[nt] = __builtin_amdgcn_wmma_f32_16x16x32_bf16(
          false, fa.v, false, fb.v, (short)0, acc[nt], false, false);
    }
  }

  // --- epilogue ---
  float beta = 0.f;
  if (EPI == 2) {
    const float s = skipv[0];
    beta = 1.f / (1.f + __expf(-s));
  }
#pragma unroll
  for (int nt = 0; nt < 8; ++nt) {
    const int col = nt * 16 + (lane & 15);
    const float b = bias[col];
#pragma unroll
    for (int r = 0; r < 8; ++r) {
      const int row = rowBase + half * 8 + r;  // C layout: lanes0-15 M=r, lanes16-31 M=8+r
      if (row < M) {
        float val = acc[nt][r] + b;
        if (EPI == 1) val = fmaxf(val, 0.f);
        if (EPI == 2) val = beta * val + (1.f - beta) * xold[(size_t)row * HID + col];
        C[(size_t)row * HID + col] = val;
      }
    }
  }
}

// ---------- edge pass 1: logits + segment max ----------
// one thread per (edge, head); arel[h] (16x16) cached in LDS

__global__ __launch_bounds__(256)
void edge_logits_kernel(const float* __restrict__ ksrc,
                        const float* __restrict__ qdst,
                        const float* __restrict__ arel,
                        const float* __restrict__ prel,
                        const int* __restrict__ esrc,
                        const int* __restrict__ edst,
                        int E,
                        float* __restrict__ logits,
                        float* __restrict__ amax)
{
  __shared__ float sA[NH * HD * HD];
  __shared__ float sp[NH];
  for (int i = threadIdx.x; i < NH * HD * HD; i += 256) sA[i] = arel[i];
  if (threadIdx.x < NH) sp[threadIdx.x] = prel[threadIdx.x];
  __syncthreads();

  const long long idx = (long long)blockIdx.x * 256 + threadIdx.x;
  if (idx >= (long long)E * NH) return;
  const int e = (int)(idx >> 3);
  const int h = (int)(idx & 7);
  const int s = esrc[e];
  const int d = edst[e];

  float kv[16], qv[16];
  load16f(ksrc + (size_t)s * HID + h * HD, kv);
  load16f(qdst + (size_t)d * HID + h * HD, qv);

  const float* Ah = sA + h * (HD * HD);
  float acc = 0.f;
#pragma unroll
  for (int f = 0; f < 16; ++f) {
    float ke = 0.f;
#pragma unroll
    for (int dd = 0; dd < 16; ++dd) ke = fmaf(kv[dd], Ah[dd * 16 + f], ke);
    acc = fmaf(qv[f], ke, acc);
  }
  const float lg = acc * sp[h] * 0.25f;  // scale = 1/sqrt(16)
  logits[(size_t)e * NH + h] = lg;
  atomicMaxFloat(&amax[(size_t)d * NH + h], lg);
}

// ---------- edge pass 2: exp / den / message accumulation ----------

__global__ __launch_bounds__(256)
void edge_accum_kernel(const float* __restrict__ vsrc,
                       const float* __restrict__ mrel,
                       const int* __restrict__ esrc,
                       const int* __restrict__ edst,
                       int E,
                       const float* __restrict__ logits,
                       const float* __restrict__ amax,
                       float* __restrict__ den,
                       float* __restrict__ msg)
{
  __shared__ float sM[NH * HD * HD];
  for (int i = threadIdx.x; i < NH * HD * HD; i += 256) sM[i] = mrel[i];
  __syncthreads();

  const long long idx = (long long)blockIdx.x * 256 + threadIdx.x;
  if (idx >= (long long)E * NH) return;
  const int e = (int)(idx >> 3);
  const int h = (int)(idx & 7);
  const int s = esrc[e];
  const int d = edst[e];

  float vv[16];
  load16f(vsrc + (size_t)s * HID + h * HD, vv);

  const float* Mh = sM + h * (HD * HD);
  float ve[16];
#pragma unroll
  for (int f = 0; f < 16; ++f) {
    float a = 0.f;
#pragma unroll
    for (int dd = 0; dd < 16; ++dd) a = fmaf(vv[dd], Mh[dd * 16 + f], a);
    ve[f] = a;
  }

  const float w = __expf(logits[(size_t)e * NH + h] - amax[(size_t)d * NH + h]);
  atomicAdd(&den[(size_t)d * NH + h], w);
  float* mp = msg + (size_t)d * HID + h * HD;
#pragma unroll
  for (int f = 0; f < 16; ++f) atomicAdd(&mp[f], w * ve[f]);
}

// ---------- normalize by den + tanh-GELU (in place on msg) ----------

__global__ __launch_bounds__(256)
void norm_gelu_kernel(float* __restrict__ msg, const float* __restrict__ den, int N)
{
  const long long i = (long long)blockIdx.x * 256 + threadIdx.x;
  if (i >= (long long)N * HID) return;
  const int n = (int)(i >> 7);
  const int c = (int)(i & 127);
  const int h = c >> 4;
  const float dd = fmaxf(den[(size_t)n * NH + h], 1e-16f);
  const float x = msg[i] / dd;
  const float x3 = x * x * x;
  const float g = 0.5f * x * (1.f + tanhf(0.7978845608028654f * (x + 0.044715f * x3)));
  msg[i] = g;
}

// ---------- fill ----------

__global__ __launch_bounds__(256)
void fill_kernel(float* __restrict__ p, float v, long long n)
{
  const long long i = (long long)blockIdx.x * 256 + threadIdx.x;
  if (i < n) p[i] = v;
}

// ---------- classifier: out[n,c] = dot(x[n], Wcls[c]) + bcls[c] ----------

__global__ __launch_bounds__(256)
void cls_kernel(const float* __restrict__ x, const float* __restrict__ Wc,
                const float* __restrict__ bc, float* __restrict__ out, int N)
{
  __shared__ float sW[4 * HID];
  __shared__ float sb[4];
  for (int i = threadIdx.x; i < 4 * HID; i += 256) sW[i] = Wc[i];
  if (threadIdx.x < 4) sb[threadIdx.x] = bc[threadIdx.x];
  __syncthreads();

  const long long i = (long long)blockIdx.x * 256 + threadIdx.x;
  if (i >= (long long)N * 4) return;
  const int n = (int)(i >> 2);
  const int c = (int)(i & 3);
  const float* xp = x + (size_t)n * HID;
  const float* wp = sW + c * HID;
  float acc = 0.f;
#pragma unroll 8
  for (int d = 0; d < HID; ++d) acc = fmaf(xp[d], wp[d], acc);
  out[i] = acc + sb[c];
}

// ---------- orchestration ----------

extern "C" void kernel_launch(void* const* d_in, const int* in_sizes, int n_in,
                              void* d_out, int out_size, void* d_ws, size_t ws_size,
                              hipStream_t stream)
{
  const float* x_a  = (const float*)d_in[0];
  const float* x_b  = (const float*)d_in[1];
  const float* lWa  = (const float*)d_in[2];
  const float* lba  = (const float*)d_in[3];
  const float* lWb  = (const float*)d_in[4];
  const float* lbb  = (const float*)d_in[5];
  const float* Wk   = (const float*)d_in[6];
  const float* bk   = (const float*)d_in[7];
  const float* Wq   = (const float*)d_in[8];
  const float* bq   = (const float*)d_in[9];
  const float* Wv   = (const float*)d_in[10];
  const float* bv   = (const float*)d_in[11];
  const float* arel = (const float*)d_in[12];
  const float* mrel = (const float*)d_in[13];
  const float* prel = (const float*)d_in[14];
  const float* Wa   = (const float*)d_in[15];
  const float* ba   = (const float*)d_in[16];
  const float* skip = (const float*)d_in[17];
  const float* Wcls = (const float*)d_in[18];
  const float* bcls = (const float*)d_in[19];
  const int* edges[4] = {(const int*)d_in[20], (const int*)d_in[21],
                         (const int*)d_in[22], (const int*)d_in[23]};
  int Esz[4] = {in_sizes[20] / 2, in_sizes[21] / 2, in_sizes[22] / 2, in_sizes[23] / 2};

  const int Na = in_sizes[0] / 64;
  const int Nb = in_sizes[1] / 32;
  int N[2] = {Na, Nb};
  const int Nmax = Na > Nb ? Na : Nb;
  int Emax = 0;
  for (int i = 0; i < 4; ++i) if (Esz[i] > Emax) Emax = Esz[i];

  // workspace layout
  char* ws = (char*)d_ws;
  size_t off = 0;
  auto alloc = [&](size_t elems) -> float* {
    float* p = (float*)(ws + off);
    off += (elems * sizeof(float) + 255) & ~(size_t)255;
    return p;
  };
  float* x[2]  = {alloc((size_t)Na * HID), alloc((size_t)Nb * HID)};
  float* xn[2] = {alloc((size_t)Na * HID), alloc((size_t)Nb * HID)};
  float* kb[2] = {alloc((size_t)Na * HID), alloc((size_t)Nb * HID)};
  float* qb[2] = {alloc((size_t)Na * HID), alloc((size_t)Nb * HID)};
  float* vb[2] = {alloc((size_t)Na * HID), alloc((size_t)Nb * HID)};
  float* msg   = alloc((size_t)Nmax * HID);
  float* den   = alloc((size_t)Nmax * NH);
  float* amax  = alloc((size_t)Nmax * NH);
  float* lg[2] = {alloc((size_t)Emax * NH), alloc((size_t)Emax * NH)};

  const dim3 blk(256);
  auto gb = [](long long n) { return dim3((unsigned)((n + 255) / 256)); };

  // input projections + ReLU
  gemm_wmma_kernel<64, 1><<<dim3((Na + 127) / 128), blk, 0, stream>>>(
      x_a, lWa, lba, nullptr, nullptr, x[0], Na);
  gemm_wmma_kernel<32, 1><<<dim3((Nb + 127) / 128), blk, 0, stream>>>(
      x_b, lWb, lbb, nullptr, nullptr, x[1], Nb);

  // relations feeding each dst type: (relation idx, src type)
  const int relR[2][2]  = {{0, 2}, {1, 3}};
  const int relSt[2][2] = {{0, 1}, {0, 1}};

  for (int l = 0; l < 2; ++l) {
    // K/Q/V projections per type
    for (int t = 0; t < 2; ++t) {
      const size_t wo = (size_t)(l * 2 + t) * HID * HID;
      const size_t bo = (size_t)(l * 2 + t) * HID;
      const dim3 g((N[t] + 127) / 128);
      gemm_wmma_kernel<128, 0><<<g, blk, 0, stream>>>(x[t], Wk + wo, bk + bo, nullptr, nullptr, kb[t], N[t]);
      gemm_wmma_kernel<128, 0><<<g, blk, 0, stream>>>(x[t], Wq + wo, bq + bo, nullptr, nullptr, qb[t], N[t]);
      gemm_wmma_kernel<128, 0><<<g, blk, 0, stream>>>(x[t], Wv + wo, bv + bo, nullptr, nullptr, vb[t], N[t]);
    }

    for (int dt = 0; dt < 2; ++dt) {
      fill_kernel<<<gb((long long)N[dt] * NH), blk, 0, stream>>>(amax, -3.402823466e38f, (long long)N[dt] * NH);
      fill_kernel<<<gb((long long)N[dt] * NH), blk, 0, stream>>>(den, 0.f, (long long)N[dt] * NH);
      fill_kernel<<<gb((long long)N[dt] * HID), blk, 0, stream>>>(msg, 0.f, (long long)N[dt] * HID);

      for (int j = 0; j < 2; ++j) {
        const int r = relR[dt][j], st = relSt[dt][j];
        const int E = Esz[r];
        edge_logits_kernel<<<gb((long long)E * NH), blk, 0, stream>>>(
            kb[st], qb[dt],
            arel + (size_t)(l * 4 + r) * NH * HD * HD,
            prel + (size_t)(l * 4 + r) * NH,
            edges[r], edges[r] + E, E, lg[j], amax);
      }
      for (int j = 0; j < 2; ++j) {
        const int r = relR[dt][j], st = relSt[dt][j];
        const int E = Esz[r];
        edge_accum_kernel<<<gb((long long)E * NH), blk, 0, stream>>>(
            vb[st],
            mrel + (size_t)(l * 4 + r) * NH * HD * HD,
            edges[r], edges[r] + E, E, lg[j], amax, den, msg);
      }

      norm_gelu_kernel<<<gb((long long)N[dt] * HID), blk, 0, stream>>>(msg, den, N[dt]);

      const size_t wo = (size_t)(l * 2 + dt) * HID * HID;
      const size_t bo = (size_t)(l * 2 + dt) * HID;
      gemm_wmma_kernel<128, 2><<<dim3((N[dt] + 127) / 128), blk, 0, stream>>>(
          msg, Wa + wo, ba + bo, x[dt], skip + (l * 2 + dt), xn[dt], N[dt]);
    }

    // ping-pong x buffers
    float* t0 = x[0]; float* t1 = x[1];
    x[0] = xn[0]; x[1] = xn[1];
    xn[0] = t0;  xn[1] = t1;
  }

  cls_kernel<<<gb((long long)Na * 4), blk, 0, stream>>>(x[0], Wcls, bcls, (float*)d_out, Na);

  (void)n_in; (void)out_size; (void)ws_size;
}
